// WGAAE_12463995093544
// MI455X (gfx1250) — matrix operand order repaired and checked by reading, not compiled, not run on hardware
//
#include <hip/hip_runtime.h>
#include <hip/hip_bf16.h>
#include <math.h>

#define N_NODES  20000
#define N_EDGES  640000
#define EE_EDGES (N_EDGES + N_NODES)
#define HEADS    4
#define REAL_MINF 2.2e-10f
#define SAMPLE_NUM 10

typedef __attribute__((ext_vector_type(16))) __bf16 v16bf;
typedef __attribute__((ext_vector_type(8)))  float  v8f;

union FragBF { v16bf v; uint4 q[2]; };

__device__ __forceinline__ float softplus_f(float x) {
  return (x > 20.0f) ? x : log1pf(expf(x));
}

__device__ __forceinline__ unsigned short f2bf(float f) {
  unsigned int b = __float_as_uint(f);
  unsigned int r = b + 0x7FFFu + ((b >> 16) & 1u);   // round-to-nearest-even
  return (unsigned short)(r >> 16);
}

__global__ void k_f32_to_bf16(const float* __restrict__ in,
                              unsigned short* __restrict__ out, int n) {
  for (int i = blockIdx.x * blockDim.x + threadIdx.x; i < n;
       i += gridDim.x * blockDim.x)
    out[i] = f2bf(in[i]);
}

// Weight convert + transpose: W[K,N] f32 -> Bt[N,K] bf16 (K-contiguous rows)
__global__ void k_w_bf16_T(const float* __restrict__ W,
                           unsigned short* __restrict__ Bt, int K, int N) {
  for (int i = blockIdx.x * blockDim.x + threadIdx.x; i < K * N;
       i += gridDim.x * blockDim.x) {
    int k = i / N, n = i - k * N;
    Bt[(size_t)n * K + k] = f2bf(W[i]);
  }
}

__global__ void k_fill_f32(float* __restrict__ p, float v, int n) {
  for (int i = blockIdx.x * blockDim.x + threadIdx.x; i < n;
       i += gridDim.x * blockDim.x)
    p[i] = v;
}

// ---------------------------------------------------------------------------
// Tiled bf16 WMMA GEMM: C[M,N] = A[M,K] @ Bt^T (+bias)(+addsrc)(softplus)
//   A : [M,K] bf16 row-major,  Bt : [N,K] bf16 row-major (pre-transposed)
// Block = 256 threads = 8 wave32, tiled 4(M) x 2(N); each wave owns a 16x32
// output tile -> 2 x v_wmma_f32_16x16x32_bf16 per BK=32 step. Block tile 64x64.
// Staging: one async b128 mem->LDS transfer per thread per tile (ASYNCcnt),
// K-tail chunks zero-filled, M-tail rows clamped (discarded in epilogue).
// Requires: N % 64 == 0, K % 8 == 0 (true for every GEMM in this model).
// EPI bits: 1 = +bias[col], 2 = +addsrc[row,col], 4 = softplus
// ---------------------------------------------------------------------------
template <int EPI>
__global__ __launch_bounds__(256) void k_gemm_wmma(
    const unsigned short* __restrict__ A,
    const unsigned short* __restrict__ Bt,
    const float* __restrict__ bias,
    const float* __restrict__ addsrc,
    float* __restrict__ C, int M, int N, int K) {
  // 40-short row stride = 80B (multiple of 16B) -> aligned b128 LDS accesses
  __shared__ __align__(16) unsigned short As[64][40];
  __shared__ __align__(16) unsigned short Bs[64][40];

  const int tid  = threadIdx.x;
  const int wave = tid >> 5, lane = tid & 31;
  const int wm = wave & 3, wn = wave >> 2;           // 4 x 2 wave grid
  const int hi = lane >> 4, lm = lane & 15;
  const int m0 = blockIdx.y * 64;
  const int n0 = blockIdx.x * 64;

  // Per-thread staging assignment: row r (0..63), 8-elem chunk q (0..3)
  const int r = tid >> 2, q = tid & 3;
  int gmA = m0 + r; if (gmA >= M) gmA = M - 1;       // clamp M tail
  const int gnB = n0 + r;                            // N % 64 == 0 -> in range
  // LDS byte offsets: generic LDS pointer's low 32 bits are the LDS address
  const unsigned ldsA = (unsigned)(size_t)&As[r][q * 8];
  const unsigned ldsB = (unsigned)(size_t)&Bs[r][q * 8];

  v8f acc0 = {}, acc1 = {};

  for (int kt = 0; kt < K; kt += 32) {
    const int gk = kt + q * 8;
    if (gk < K) {  // K % 8 == 0 -> chunk fully in-bounds
      unsigned long long ga = (unsigned long long)(size_t)(A  + (size_t)gmA * K + gk);
      unsigned long long gb = (unsigned long long)(size_t)(Bt + (size_t)gnB * K + gk);
      asm volatile("global_load_async_to_lds_b128 %0, %1, off"
                   :: "v"(ldsA), "v"(ga) : "memory");
      asm volatile("global_load_async_to_lds_b128 %0, %1, off"
                   :: "v"(ldsB), "v"(gb) : "memory");
    } else {       // K tail (K=2000): zero-fill this chunk
      uint4 z = {0u, 0u, 0u, 0u};
      *(uint4*)&As[r][q * 8] = z;
      *(uint4*)&Bs[r][q * 8] = z;
    }
    asm volatile("s_wait_asynccnt 0x0" ::: "memory");
    __syncthreads();

    // Lane fragments per CDNA5 WMMA layouts:
    // A 16x32: halves 0..7 = K hi*8+0..7 ; halves 8..15 = K 16+hi*8+0..7
    // B 32x16 (from K-contiguous Bs rows): same K order as A, col = lm
    FragBF fa, fb0, fb1;
    fa.q[0]  = *(const uint4*)&As[wm * 16 + lm][hi * 8];
    fa.q[1]  = *(const uint4*)&As[wm * 16 + lm][16 + hi * 8];
    fb0.q[0] = *(const uint4*)&Bs[wn * 32 + lm][hi * 8];
    fb0.q[1] = *(const uint4*)&Bs[wn * 32 + lm][16 + hi * 8];
    fb1.q[0] = *(const uint4*)&Bs[wn * 32 + 16 + lm][hi * 8];
    fb1.q[1] = *(const uint4*)&Bs[wn * 32 + 16 + lm][16 + hi * 8];

    acc0 = __builtin_amdgcn_wmma_f32_16x16x32_bf16(
        false, fa.v, false, fb0.v, (short)0, acc0, false, false);
    acc1 = __builtin_amdgcn_wmma_f32_16x16x32_bf16(
        false, fa.v, false, fb1.v, (short)0, acc1, false, false);
    __syncthreads();
  }

  // C/D layout: VGPR idx p -> row = p + 8*hi within the wave tile, col = lm
  v8f accs[2] = {acc0, acc1};
  for (int cs = 0; cs < 2; ++cs) {
    const int col = n0 + wn * 32 + cs * 16 + lm;
    if (col < N) {
      for (int p = 0; p < 8; ++p) {
        int row = m0 + wm * 16 + hi * 8 + p;
        if (row < M) {
          float v = accs[cs][p];
          if (EPI & 1) v += bias[col];
          if (EPI & 2) v += addsrc[(size_t)row * N + col];
          if (EPI & 4) v = softplus_f(v);
          C[(size_t)row * N + col] = v;
        }
      }
    }
  }
}

// ---------------------------------------------------------------------------
// GAT attention pieces
// ---------------------------------------------------------------------------
__global__ void k_alpha(const float* __restrict__ xp,
                        const float* __restrict__ asrc,
                        const float* __restrict__ adst,
                        float* __restrict__ as_, float* __restrict__ ad_,
                        int n, int dout, int oh) {
  int i = blockIdx.x * blockDim.x + threadIdx.x;
  if (i >= n * HEADS) return;
  int node = i >> 2, h = i & 3;
  const float* xr = xp + (size_t)node * dout + h * oh;
  const float* a1 = asrc + h * oh;
  const float* a2 = adst + h * oh;
  float s = 0.0f, d = 0.0f;
  for (int j = 0; j < oh; ++j) { float v = xr[j]; s += v * a1[j]; d += v * a2[j]; }
  as_[i] = s;
  ad_[i] = d;
}

__device__ __forceinline__ void atomicMaxF(float* addr, float val) {
  int iv = __float_as_int(val);
  if (iv >= 0) atomicMax((int*)addr, iv);
  else         atomicMin((unsigned int*)addr, (unsigned int)iv);
}

__global__ void k_edge_max(const int* __restrict__ src, const int* __restrict__ dst,
                           const float* __restrict__ as_, const float* __restrict__ ad_,
                           float* __restrict__ e, float* __restrict__ emax) {
  int i = blockIdx.x * blockDim.x + threadIdx.x;
  if (i >= EE_EDGES * HEADS) return;
  int eid = i >> 2, h = i & 3;
  int s = (eid < N_EDGES) ? src[eid] : (eid - N_EDGES);  // self loops appended
  int d = (eid < N_EDGES) ? dst[eid] : (eid - N_EDGES);
  float v = as_[s * HEADS + h] + ad_[d * HEADS + h];
  v = (v > 0.0f) ? v : 0.2f * v;                         // leaky_relu(0.2)
  e[i] = v;
  atomicMaxF(&emax[d * HEADS + h], v);
}

__global__ void k_edge_expsum(const int* __restrict__ dst,
                              const float* __restrict__ e,
                              const float* __restrict__ emax,
                              float* __restrict__ ex, float* __restrict__ denom) {
  int i = blockIdx.x * blockDim.x + threadIdx.x;
  if (i >= EE_EDGES * HEADS) return;
  int eid = i >> 2, h = i & 3;
  int d = (eid < N_EDGES) ? dst[eid] : (eid - N_EDGES);
  float mx = emax[d * HEADS + h];
  if (mx < -1e29f) mx = 0.0f;                            // where(isfinite, ., 0)
  float v = expf(e[i] - mx);
  ex[i] = v;
  atomicAdd(&denom[d * HEADS + h], v);
}

__global__ void k_edge_aggregate(const int* __restrict__ src, const int* __restrict__ dst,
                                 const float* __restrict__ ex, const float* __restrict__ denom,
                                 const float* __restrict__ xp, float* __restrict__ gagg,
                                 int dout, int oh) {
  int eid = blockIdx.x;
  int s = (eid < N_EDGES) ? src[eid] : (eid - N_EDGES);
  int d = (eid < N_EDGES) ? dst[eid] : (eid - N_EDGES);
  for (int f = threadIdx.x; f < dout; f += blockDim.x) {
    int h = f / oh;
    float att = ex[eid * HEADS + h] / fmaxf(denom[d * HEADS + h], 1e-16f);
    atomicAdd(&gagg[(size_t)d * dout + f], att * xp[(size_t)s * dout + f]);
  }
}

// g = gagg + bias; also emit bf16 copy (WMMA input for the skip GEMM)
__global__ void k_bias_dual(const float* __restrict__ in, const float* __restrict__ bias,
                            float* __restrict__ outf, unsigned short* __restrict__ outbf,
                            int n, int cols) {
  for (int i = blockIdx.x * blockDim.x + threadIdx.x; i < n;
       i += gridDim.x * blockDim.x) {
    float v = in[i] + bias[i % cols];
    outf[i]  = v;
    outbf[i] = f2bf(v);
  }
}

// ---------------------------------------------------------------------------
// BatchNorm (training normalization: biased var)
// ---------------------------------------------------------------------------
__global__ __launch_bounds__(256) void k_bn_stats(const float* __restrict__ g,
                                                  float* __restrict__ stats,
                                                  int rows, int cols) {
  __shared__ float s1[256], s2[256];
  int c = blockIdx.x;
  float a = 0.0f, b = 0.0f;
  for (int r = threadIdx.x; r < rows; r += 256) {
    float v = g[(size_t)r * cols + c];
    a += v; b += v * v;
  }
  s1[threadIdx.x] = a; s2[threadIdx.x] = b;
  __syncthreads();
  for (int s = 128; s > 0; s >>= 1) {
    if (threadIdx.x < s) {
      s1[threadIdx.x] += s1[threadIdx.x + s];
      s2[threadIdx.x] += s2[threadIdx.x + s];
    }
    __syncthreads();
  }
  if (threadIdx.x == 0) {
    float m = s1[0] / rows;
    stats[c]        = m;
    stats[cols + c] = s2[0] / rows - m * m;
  }
}

__global__ void k_bn_apply(const float* __restrict__ in, const float* __restrict__ stats,
                           const float* __restrict__ gamma, const float* __restrict__ beta,
                           float* __restrict__ outf, unsigned short* __restrict__ outbf,
                           int n, int cols) {
  for (int i = blockIdx.x * blockDim.x + threadIdx.x; i < n;
       i += gridDim.x * blockDim.x) {
    int c = i % cols;
    float v = gamma[c] * (in[i] - stats[c]) * rsqrtf(stats[cols + c] + 1e-5f) + beta[c];
    outf[i]  = v;
    outbf[i] = f2bf(v);   // bf16 input for next layer / decoder GEMM
  }
}

// ---------------------------------------------------------------------------
// Weibull reparameterized sampling (hash RNG), output [Kt, N] row-major
// ---------------------------------------------------------------------------
__device__ __forceinline__ float hash_u01(unsigned int x) {
  x ^= x >> 17; x *= 0xED5AD4BBu;
  x ^= x >> 11; x *= 0xAC4C1B51u;
  x ^= x >> 15; x *= 0x31848BABu;
  x ^= x >> 14;
  return (float)(x >> 8) * (1.0f / 16777216.0f);
}

__global__ void k_theta(const float* __restrict__ kmat,   // [N, Kt]
                        const float* __restrict__ lmat,   // [N, Kt]
                        float* __restrict__ out,          // [Kt, N]
                        int Kt, int n, int t) {
  int i = blockIdx.x * blockDim.x + threadIdx.x;
  if (i >= Kt * n) return;
  int kk = i / n, node = i - kk * n;
  size_t idx = (size_t)node * Kt + kk;
  float k = fminf(fmaxf(kmat[idx], 0.1f), 1000.0f);
  float l = fmaxf(lmat[idx], REAL_MINF);
  l = l / expf(lgammaf(1.0f + 1.0f / k));
  float invk = 1.0f / k;
  float accv = 0.0f;
  unsigned int base = ((unsigned int)t * 0x9E3779B9u) ^ ((unsigned int)i * 0x85EBCA6Bu);
  for (int s = 0; s < SAMPLE_NUM; ++s) {
    float u  = hash_u01(base + (unsigned int)s * 0xC2B2AE35u + 0x165667B1u);
    float om = fmaxf(1.0f - u, REAL_MINF);
    accv += powf(-logf(om), invk);
  }
  float th = l * accv * (1.0f / (float)SAMPLE_NUM);
  out[i] = fminf(fmaxf(th, REAL_MINF), 1000.0f);
}

// ---------------------------------------------------------------------------
// Host launcher
// ---------------------------------------------------------------------------
extern "C" void kernel_launch(void* const* d_in, const int* in_sizes, int n_in,
                              void* d_out, int out_size, void* d_ws, size_t ws_size,
                              hipStream_t stream) {
  (void)in_sizes; (void)n_in; (void)out_size; (void)ws_size;

  const float* x  = (const float*)d_in[0];
  const int*   ei = (const int*)d_in[1];
  const float *gat_W[3], *gat_asrc[3], *gat_adst[3], *gat_b[3];
  const float *skip_W[3], *skip_b[3], *bn_g[3], *bn_b[3];
  const float *mlp_W[3], *mlp_b[3], *shp_W[3], *shp_b[3], *scl_W[3], *scl_b[3];
  int idx = 2;
  auto take3 = [&](const float** a) {
    for (int j = 0; j < 3; ++j) a[j] = (const float*)d_in[idx++];
  };
  take3(gat_W); take3(gat_asrc); take3(gat_adst); take3(gat_b);
  take3(skip_W); take3(skip_b); take3(bn_g); take3(bn_b);
  take3(mlp_W); take3(mlp_b); take3(shp_W); take3(shp_b);
  take3(scl_W); take3(scl_b);

  const int HDIM[4] = {2000, 512, 256, 128};
  const int KT[3]   = {256, 128, 64};
  const int N = N_NODES, E = N_EDGES, EE = EE_EDGES;
  const int* srcA = ei;
  const int* dstA = ei + E;

  // Workspace carve-out
  char* wsb = (char*)d_ws;
  size_t off = 0;
  auto alloc = [&](size_t bytes) -> char* {
    char* p = wsb + off;
    off += (bytes + 255) & ~(size_t)255;
    return p;
  };
  unsigned short* bfIn = (unsigned short*)alloc((size_t)N * 2000 * 2); // act bf16
  unsigned short* bfW  = (unsigned short*)alloc((size_t)2000 * 512 * 2); // W^T bf16
  float* fX    = (float*)alloc((size_t)N * 512 * 4);   // xp / hm
  float* fG    = (float*)alloc((size_t)N * 512 * 4);   // gagg / g2 / kmat
  float* fH    = (float*)alloc((size_t)N * 512 * 4);   // g / lmat
  float* eBuf  = (float*)alloc((size_t)EE * HEADS * 4);
  float* exBuf = (float*)alloc((size_t)EE * HEADS * 4);
  float* emax  = (float*)alloc((size_t)N * HEADS * 4);
  float* denom = (float*)alloc((size_t)N * HEADS * 4);
  float* alphaS = (float*)alloc((size_t)N * HEADS * 4);
  float* alphaD = (float*)alloc((size_t)N * HEADS * 4);
  float* stats = (float*)alloc(2 * 512 * 4);
  float* hsBuf = (float*)alloc((size_t)N * (512 + 256 + 128) * 4);
  const size_t hsOff[3] = {0, (size_t)N * 512, (size_t)N * (512 + 256)};

  auto cvt = [&](const float* s, unsigned short* d, size_t n) {
    size_t b = (n + 255) / 256; if (b > 4096) b = 4096;
    k_f32_to_bf16<<<(unsigned)b, 256, 0, stream>>>(s, d, (int)n);
  };
  auto cvtT = [&](const float* s, unsigned short* d, int K, int Ncols) {
    size_t n = (size_t)K * Ncols;
    size_t b = (n + 255) / 256; if (b > 4096) b = 4096;
    k_w_bf16_T<<<(unsigned)b, 256, 0, stream>>>(s, d, K, Ncols);
  };
  auto fill = [&](float* p, float v, int n) {
    int b = (n + 255) / 256; if (b > 8192) b = 8192;
    k_fill_f32<<<b, 256, 0, stream>>>(p, v, n);
  };

  // ---------------- Encoder ----------------
  cvt(x, bfIn, (size_t)N * HDIM[0]);
  for (int t = 0; t < 3; ++t) {
    const int din = HDIM[t], dout = HDIM[t + 1], oh = dout / HEADS;
    dim3 gg((dout + 63) / 64, (N + 63) / 64);

    // xp = h @ W (no bias yet, matches reference)
    cvtT(gat_W[t], bfW, din, dout);
    k_gemm_wmma<0><<<gg, 256, 0, stream>>>(bfIn, bfW, nullptr, nullptr, fX, N, dout, din);

    // per-node attention logits
    k_alpha<<<(N * HEADS + 255) / 256, 256, 0, stream>>>(
        fX, gat_asrc[t], gat_adst[t], alphaS, alphaD, N, dout, oh);

    // edge softmax + aggregation
    fill(emax, -1e30f, N * HEADS);
    fill(denom, 0.0f, N * HEADS);
    fill(fG, 0.0f, N * dout);
    int eb = (EE * HEADS + 255) / 256;
    k_edge_max   <<<eb, 256, 0, stream>>>(srcA, dstA, alphaS, alphaD, eBuf, emax);
    k_edge_expsum<<<eb, 256, 0, stream>>>(dstA, eBuf, emax, exBuf, denom);
    k_edge_aggregate<<<EE, 128, 0, stream>>>(srcA, dstA, exBuf, denom, fX, fG, dout, oh);

    // g = agg + gat_b  (f32 -> fH, bf16 -> bfIn for skip GEMM)
    k_bias_dual<<<(N * dout + 255) / 256, 256, 0, stream>>>(
        fG, gat_b[t], fH, bfIn, N * dout, dout);

    // g2 = g + g @ skip_W + skip_b  (fused residual epilogue)
    cvtT(skip_W[t], bfW, dout, dout);
    k_gemm_wmma<3><<<gg, 256, 0, stream>>>(bfIn, bfW, skip_b[t], fH, fG, N, dout, dout);

    // BatchNorm -> hs[t] (f32) and next-layer input (bf16)
    k_bn_stats<<<dout, 256, 0, stream>>>(fG, stats, N, dout);
    k_bn_apply<<<(N * dout + 255) / 256, 256, 0, stream>>>(
        fG, stats, bn_g[t], bn_b[t], hsBuf + hsOff[t], bfIn, N * dout, dout);
  }

  // ---------------- Decoder ----------------
  float* outp = (float*)d_out;
  size_t outOff = 0;
  for (int t = 0; t < 3; ++t) {
    const int dout = HDIM[t + 1], Kt = KT[t];
    dim3 g1((Kt + 63) / 64, (N + 63) / 64);

    cvt(hsBuf + hsOff[t], bfIn, (size_t)N * dout);
    cvtT(mlp_W[t], bfW, dout, Kt);
    k_gemm_wmma<5><<<g1, 256, 0, stream>>>(bfIn, bfW, mlp_b[t], nullptr, fX, N, Kt, dout); // hm

    cvt(fX, bfIn, (size_t)N * Kt);            // hm -> bf16
    cvtT(shp_W[t], bfW, Kt, Kt);
    k_gemm_wmma<5><<<g1, 256, 0, stream>>>(bfIn, bfW, shp_b[t], nullptr, fG, N, Kt, Kt);   // k
    cvtT(scl_W[t], bfW, Kt, Kt);
    k_gemm_wmma<5><<<g1, 256, 0, stream>>>(bfIn, bfW, scl_b[t], nullptr, fH, N, Kt, Kt);   // l

    k_theta<<<(Kt * N + 255) / 256, 256, 0, stream>>>(fG, fH, outp + outOff, Kt, N, t);
    outOff += (size_t)Kt * N;
  }
}
